// Tox21Net_21277267984764
// MI455X (gfx1250) — compile-verified, matrix operand order checked by reading, not run on hardware
//
#include <hip/hip_runtime.h>
#include <hip/hip_bf16.h>

typedef __attribute__((ext_vector_type(8)))  _Float16 v8h;
typedef __attribute__((ext_vector_type(16))) _Float16 v16h;
typedef __attribute__((ext_vector_type(8)))  float    v8f;

#define N_NODES  100000
#define N_EDGES  1600000
#define N_GRAPHS 2048
#define DIM      75
#define ZIN      154
#define KPAD     160   // padded z length (5 WMMA K-steps of 32)
#define NPAD     80    // padded output dim (5 WMMA N-tiles of 16)
#define H16LD    80    // padded f16 node row

__device__ __forceinline__ float fast_rcp(float x) { return __builtin_amdgcn_rcpf(x); }
__device__ __forceinline__ float sigmoidf_(float x) {
    return fast_rcp(1.f + __expf(-x));          // v_exp_f32 + v_rcp_f32
}
__device__ __forceinline__ float softplusf_(float x) {
    // max(x,0) + log(1 + exp(-|x|))  (stable softplus)
    return fmaxf(x, 0.f) + __logf(1.f + __expf(-fabsf(x)));
}

// ---------------------------------------------------------------------------
// Repack wf/ws (f32 [75][154]) into padded f16 [80][160] matching z layout:
//   cols 0..74  = w[:, 0..74]   (dst-half)
//   cols 75..79 = 0
//   cols 80..158= w[:, 75..153] (src-half + edge attrs)
//   col  159    = 0
// Output: wp[0 .. 80*160) = wf_packed,  wp[80*160 ..) = ws_packed
// ---------------------------------------------------------------------------
__global__ void pack_weights_kernel(const float* __restrict__ wf,
                                    const float* __restrict__ ws,
                                    _Float16* __restrict__ wp) {
    int i = blockIdx.x * blockDim.x + threadIdx.x;
    if (i >= NPAD * KPAD) return;
    int n = i / KPAD, c = i % KPAD;
    float vf = 0.f, vs = 0.f;
    if (n < DIM) {
        int s = -1;
        if (c < DIM) s = c;
        else if (c >= 80 && c < 159) s = c - 5;
        if (s >= 0) { vf = wf[n * ZIN + s]; vs = ws[n * ZIN + s]; }
    }
    wp[i] = (_Float16)vf;
    wp[NPAD * KPAD + i] = (_Float16)vs;
}

// h = relu(x @ w0.T + b0) : [N,53] -> [N,75]
__global__ void embed_kernel(const float* __restrict__ x, const float* __restrict__ w0,
                             const float* __restrict__ b0, float* __restrict__ h) {
    __shared__ float sx[53];
    int node = blockIdx.x, t = threadIdx.x;
    if (t < 53) sx[t] = x[node * 53 + t];
    __syncthreads();
    if (t < DIM) {
        float acc = b0[t];
        const float* wr = w0 + t * 53;
        #pragma unroll
        for (int k = 0; k < 53; ++k) acc = fmaf(sx[k], wr[k], acc);
        h[node * DIM + t] = fmaxf(acc, 0.f);
    }
}

// f16 mirror of h (padded row of 80, cols 75..79 = 0) + zero agg
__global__ void prep_kernel(const float* __restrict__ h, _Float16* __restrict__ h16,
                            float* __restrict__ agg) {
    int i = blockIdx.x * blockDim.x + threadIdx.x;
    if (i >= N_NODES * H16LD) return;
    int node = i / H16LD, c = i % H16LD;
    float v = (c < DIM) ? h[node * DIM + c] : 0.f;
    h16[i] = (_Float16)v;
    if (c < DIM) agg[node * DIM + c] = 0.f;
}

// ---------------------------------------------------------------------------
// Edge MLP: per 16-edge tile, z[16x160]f16 @ {wf,ws}^T[160x80]f16 -> f32,
// sigmoid*softplus epilogue, atomic scatter into agg[dst].
// 2 waves/WG, one tile per wave; B fragments re-read from LDS every tile
// (memory fence blocks LICM so the wave stays ~160 VGPRs for occupancy).
// LDS: 51200B weights + 2*5120B z tiles + 128B dst cache = 61568B.
// ---------------------------------------------------------------------------
__global__ __launch_bounds__(64) void edge_mlp_kernel(
    const _Float16* __restrict__ h16, const _Float16* __restrict__ wp,
    const float* __restrict__ bf, const float* __restrict__ bs,
    const int* __restrict__ srcIdx, const int* __restrict__ dstIdx,
    const float* __restrict__ eattr, float* __restrict__ agg) {

    __shared__ __align__(16) _Float16 sW[2 * NPAD * KPAD];   // wf then ws
    __shared__ __align__(16) _Float16 sZ[2][16 * KPAD];      // per-wave z tile
    __shared__ int sDst[2][16];                              // per-wave dst cache

    // stage packed weights once per workgroup
    {
        const float4* s = (const float4*)wp;
        float4* d = (float4*)sW;
        for (int i = threadIdx.x; i < (2 * NPAD * KPAD) / 8; i += blockDim.x) d[i] = s[i];
    }
    __syncthreads();

    const int lane = threadIdx.x & 31;
    const int wave = threadIdx.x >> 5;
    const int numWaves = gridDim.x * 2;
    const int wgWave = blockIdx.x * 2 + wave;

    _Float16* zt = sZ[wave];
    const v8h* zv = (const v8h*)zt;       // z tile, 20 v8h per row
    const v8h* wv = (const v8h*)sW;       // weights, 20 v8h per row

    const int mrow = lane & 15;           // A row / C column select
    const int colB = lane & 15;           // B column
    const int ksel = (lane >> 4) & 1;     // K-chunk select (ISA 16-bit A/B layout)

    // hoist per-lane biases out of the tile loop (fence below would reload them)
    float bfl[5], bsl[5];
    #pragma unroll
    for (int n = 0; n < 5; ++n) {
        int col = n * 16 + colB;
        bfl[n] = (col < DIM) ? bf[col] : 0.f;
        bsl[n] = (col < DIM) ? bs[col] : 0.f;
    }

    const v8f vzero = {0.f, 0.f, 0.f, 0.f, 0.f, 0.f, 0.f, 0.f};
    const int NT = N_EDGES / 16;

    for (int tile = wgWave; tile < NT; tile += numWaves) {
        // ---- stage z tile: 2 lanes per edge copy dst-row + src-row halves ----
        {
            int e = lane >> 1;
            int p = lane & 1;
            int edge = tile * 16 + e;
            int di = dstIdx[edge];
            int si = srcIdx[edge];
            if (p == 0) sDst[wave][e] = di;
            const v8h* hd = (const v8h*)(h16 + (size_t)di * H16LD) + p * 5;
            const v8h* hs = (const v8h*)(h16 + (size_t)si * H16LD) + p * 5;
            v8h* zr = (v8h*)(zt + e * KPAD) + p * 5;
            #pragma unroll
            for (int i = 0; i < 5; ++i) zr[i] = hd[i];          // cols 0..79
            #pragma unroll
            for (int i = 0; i < 5; ++i) zr[i + 10] = hs[i];     // cols 80..159
            if (lane < 16) {                                    // edge attrs -> 155..158
                float4 ea = ((const float4*)eattr)[tile * 16 + lane];
                _Float16* za = zt + lane * KPAD + 155;
                za[0] = (_Float16)ea.x; za[1] = (_Float16)ea.y;
                za[2] = (_Float16)ea.z; za[3] = (_Float16)ea.w;
            }
        }
        // Order the intra-wave LDS store->load pattern for the compiler and
        // block LICM from hoisting all 50 B fragments into VGPRs across tiles.
        asm volatile("" ::: "memory");

        v8f accF[5], accS[5];
        #pragma unroll
        for (int n = 0; n < 5; ++n) { accF[n] = vzero; accS[n] = vzero; }

        #pragma unroll
        for (int k = 0; k < 5; ++k) {
            // A fragment: lane holds row m, K chunks {0..7,16..23}/{8..15,24..31}
            int ai = mrow * 20 + k * 4 + ksel;
            v8h alo = zv[ai], ahi = zv[ai + 2];
            v16h a = __builtin_shufflevector(alo, ahi,
                0, 1, 2, 3, 4, 5, 6, 7, 8, 9, 10, 11, 12, 13, 14, 15);
            #pragma unroll
            for (int n = 0; n < 5; ++n) {
                // B fragment: lane holds column nBase+colB, same K-chunk split
                int bi = (n * 16 + colB) * 20 + k * 4 + ksel;
                v8h flo = wv[bi], fhi = wv[bi + 2];
                v8h slo = wv[bi + 1600], shi = wv[bi + 1602]; // ws at +80*160/8 v8h
                v16h bfm = __builtin_shufflevector(flo, fhi,
                    0, 1, 2, 3, 4, 5, 6, 7, 8, 9, 10, 11, 12, 13, 14, 15);
                v16h bsm = __builtin_shufflevector(slo, shi,
                    0, 1, 2, 3, 4, 5, 6, 7, 8, 9, 10, 11, 12, 13, 14, 15);
                accF[n] = __builtin_amdgcn_wmma_f32_16x16x32_f16(
                    false, a, false, bfm, (short)0, accF[n], false, false);
                accS[n] = __builtin_amdgcn_wmma_f32_16x16x32_f16(
                    false, a, false, bsm, (short)0, accS[n], false, false);
            }
        }

        // ---- epilogue: lane L, VGPR v  ->  edge m = v + 8*(L>>4), dim n*16 + (L&15)
        int mBase = (lane >> 4) * 8;
        float* aggb[8];
        #pragma unroll
        for (int v = 0; v < 8; ++v)
            aggb[v] = agg + (size_t)sDst[wave][mBase + v] * DIM;

        #pragma unroll
        for (int n = 0; n < 5; ++n) {
            int col = n * 16 + colB;
            if (col < DIM) {
                #pragma unroll
                for (int v = 0; v < 8; ++v) {
                    float g = sigmoidf_(accF[n][v] + bfl[n]);
                    float s = softplusf_(accS[n][v] + bsl[n]);
                    atomicAdd(aggb[v] + col, g * s);
                }
            }
        }
    }
}

// h = relu(h + agg)
__global__ void update_kernel(float* __restrict__ h, const float* __restrict__ agg) {
    int i = blockIdx.x * blockDim.x + threadIdx.x;
    if (i < N_NODES * DIM) h[i] = fmaxf(h[i] + agg[i], 0.f);
}

__global__ void zero_graph_kernel(float* __restrict__ gbuf) {
    int i = blockIdx.x * blockDim.x + threadIdx.x;
    if (i < N_GRAPHS * 128) gbuf[i] = 0.f;
}

// node_x = h @ w1.T + b1; positive-only atomicMax into gbuf[batch] (init 0 absorbs relu)
__global__ void proj_max_kernel(const float* __restrict__ h, const float* __restrict__ w1,
                                const float* __restrict__ b1, const int* __restrict__ batch,
                                float* __restrict__ gbuf) {
    __shared__ float sh[DIM];
    int node = blockIdx.x, t = threadIdx.x;  // 128 threads
    if (t < DIM) sh[t] = h[node * DIM + t];
    __syncthreads();
    float acc = b1[t];
    const float* wr = w1 + t * DIM;
    #pragma unroll 5
    for (int k = 0; k < DIM; ++k) acc = fmaf(sh[k], wr[k], acc);
    if (acc > 0.f) {
        int g = batch[node];
        atomicMax((int*)(gbuf + g * 128 + t), __float_as_int(acc));
    }
}

// pred = relu(g @ w2.T + b2) @ w3.T + b3
__global__ void head_kernel(const float* __restrict__ gbuf, const float* __restrict__ w2,
                            const float* __restrict__ b2, const float* __restrict__ w3,
                            const float* __restrict__ b3, float* __restrict__ out) {
    __shared__ float hid[32];
    int g = blockIdx.x, t = threadIdx.x;  // 32 threads
    const float* gr = gbuf + g * 128;
    float acc = b2[t];
    const float* wr = w2 + t * 128;
    #pragma unroll 4
    for (int k = 0; k < 128; ++k) acc = fmaf(gr[k], wr[k], acc);
    hid[t] = fmaxf(acc, 0.f);
    __syncthreads();
    if (t < 2) {
        float o = b3[t];
        const float* w = w3 + t * 32;
        #pragma unroll
        for (int k = 0; k < 32; ++k) o = fmaf(hid[k], w[k], o);
        out[g * 2 + t] = o;
    }
}

extern "C" void kernel_launch(void* const* d_in, const int* in_sizes, int n_in,
                              void* d_out, int out_size, void* d_ws, size_t ws_size,
                              hipStream_t stream) {
    (void)in_sizes; (void)n_in; (void)out_size; (void)ws_size;
    const float* x    = (const float*)d_in[0];
    const int*   eidx = (const int*)d_in[1];
    const float* ea   = (const float*)d_in[2];
    const int*   batch= (const int*)d_in[3];
    const float* w0 = (const float*)d_in[4];
    const float* b0 = (const float*)d_in[5];
    const float* wf = (const float*)d_in[6];
    const float* bf = (const float*)d_in[7];
    const float* ws = (const float*)d_in[8];
    const float* bs = (const float*)d_in[9];
    const float* w1 = (const float*)d_in[10];
    const float* b1 = (const float*)d_in[11];
    const float* w2 = (const float*)d_in[12];
    const float* b2 = (const float*)d_in[13];
    const float* w3 = (const float*)d_in[14];
    const float* b3 = (const float*)d_in[15];
    const int* srcIdx = eidx;              // edge_index[0]
    const int* dstIdx = eidx + N_EDGES;    // edge_index[1]

    char* wsb = (char*)d_ws;
    auto al = [](size_t v) { return (v + 255) & ~(size_t)255; };
    size_t off = 0;
    float*    h32  = (float*)(wsb + off);    off = al(off + (size_t)N_NODES * DIM * 4);
    _Float16* h16  = (_Float16*)(wsb + off); off = al(off + (size_t)N_NODES * H16LD * 2);
    float*    agg  = (float*)(wsb + off);    off = al(off + (size_t)N_NODES * DIM * 4);
    _Float16* wp   = (_Float16*)(wsb + off); off = al(off + (size_t)2 * NPAD * KPAD * 2);
    float*    gbuf = (float*)(wsb + off);    off = al(off + (size_t)N_GRAPHS * 128 * 4);

    pack_weights_kernel<<<(NPAD * KPAD + 255) / 256, 256, 0, stream>>>(wf, ws, wp);
    embed_kernel<<<N_NODES, 96, 0, stream>>>(x, w0, b0, h32);

    for (int layer = 0; layer < 2; ++layer) {
        prep_kernel<<<(N_NODES * H16LD + 255) / 256, 256, 0, stream>>>(h32, h16, agg);
        edge_mlp_kernel<<<4096, 64, 0, stream>>>(h16, wp, bf, bs, srcIdx, dstIdx, ea, agg);
        update_kernel<<<(N_NODES * DIM + 255) / 256, 256, 0, stream>>>(h32, agg);
    }

    zero_graph_kernel<<<(N_GRAPHS * 128 + 255) / 256, 256, 0, stream>>>(gbuf);
    proj_max_kernel<<<N_NODES, 128, 0, stream>>>(h32, w1, b1, batch, gbuf);
    head_kernel<<<N_GRAPHS, 32, 0, stream>>>(gbuf, w2, b2, w3, b3, (float*)d_out);
}
